// GPT_29429115912199
// MI455X (gfx1250) — matrix operand order repaired and checked by reading, not compiled, not run on hardware
//
#include <hip/hip_runtime.h>
#include <hip/hip_bf16.h>
#include <math.h>

// ---------------------------------------------------------------- constants
constexpr int BB = 2;
constexpr int TT = 1024;
constexpr int DD = 768;
constexpr int HH = 12;
constexpr int HD = 64;
constexpr int LL = 4;
constexpr int VV = 50257;
constexpr int VP = 50304;              // VV padded to multiple of 128

typedef __attribute__((ext_vector_type(16))) _Float16 v16h;
typedef __attribute__((ext_vector_type(8)))  _Float16 v8h;
typedef __attribute__((ext_vector_type(8)))  float    v8f;

// GEMM epilogue flags (compile-time template parameter)
#define GEMM_BIAS 1
#define GEMM_GELU 2
#define GEMM_RES  4
#define GEMM_WF32 8
#define GEMM_WF16 16

// ---------------------------------------------------------------- embedding
__global__ __launch_bounds__(256) void embed_kernel(const int* __restrict__ idx,
                                                    const float* __restrict__ wte,
                                                    float* __restrict__ x) {
  int i = blockIdx.x * 256 + threadIdx.x;   // B*T*D
  if (i >= BB * TT * DD) return;
  int d  = i % DD;
  int bt = i / DD;
  x[i] = wte[(size_t)idx[bt] * DD + d];
}

// ---------------------------------------------------------------- convert+transpose weights
// src: W [K,N] f32 (row-major)  ->  dst: W^T [Npad,K] f16, pad rows zeroed.
// K is always a multiple of 32 here. block = (32,8), grid = (Npad/32, K/32).
__global__ __launch_bounds__(256) void transpose_f16_kernel(
    const float* __restrict__ src, _Float16* __restrict__ dst,
    int K, int N, int Npad) {
  __shared__ _Float16 tile[32][33];
  const int k0 = blockIdx.y * 32;
  const int n0 = blockIdx.x * 32;
  const int tx = threadIdx.x;            // 0..31
  const int ty = threadIdx.y;            // 0..7
  #pragma unroll
  for (int i = 0; i < 32; i += 8) {
    const int k = k0 + ty + i;
    const int n = n0 + tx;
    float v = (n < N) ? src[(size_t)k * N + n] : 0.f;
    tile[tx][ty + i] = (_Float16)v;      // tile[n_local][k_local]
  }
  __syncthreads();
  #pragma unroll
  for (int i = 0; i < 32; i += 8) {
    const int n = n0 + ty + i;
    if (n < Npad) dst[(size_t)n * K + k0 + tx] = tile[ty + i][tx];
  }
}

// ---------------------------------------------------------------- dual layernorm -> f16
__global__ __launch_bounds__(256) void ln_dual_kernel(
    const float* __restrict__ x,
    const float* __restrict__ g1, const float* __restrict__ b1,
    const float* __restrict__ g2, const float* __restrict__ b2,
    _Float16* __restrict__ h1, _Float16* __restrict__ h2) {
  __shared__ float sm[2][9];
  const int row = blockIdx.x;
  const int tid = threadIdx.x;
  const float* xr = x + (size_t)row * DD;
  float s = 0.f, ss = 0.f;
  for (int i = tid; i < DD; i += 256) { float v = xr[i]; s += v; ss += v * v; }
  for (int o = 16; o > 0; o >>= 1) { s += __shfl_xor(s, o, 32); ss += __shfl_xor(ss, o, 32); }
  if ((tid & 31) == 0) { sm[0][tid >> 5] = s; sm[1][tid >> 5] = ss; }
  __syncthreads();
  if (tid == 0) {
    float a = 0.f, b = 0.f;
    for (int w = 0; w < 8; w++) { a += sm[0][w]; b += sm[1][w]; }
    sm[0][8] = a; sm[1][8] = b;
  }
  __syncthreads();
  const float mean = sm[0][8] / (float)DD;
  const float var  = sm[1][8] / (float)DD - mean * mean;
  const float rstd = rsqrtf(var + 1e-5f);
  for (int i = tid; i < DD; i += 256) {
    float v = (xr[i] - mean) * rstd;
    h1[(size_t)row * DD + i] = (_Float16)(v * g1[i] + b1[i]);
    if (h2) h2[(size_t)row * DD + i] = (_Float16)(v * g2[i] + b2[i]);
  }
}

// ---------------------------------------------------------------- partial RoPE + QKV pack
// qkv: [B,T,3D] f32 -> Qr,Kr: [B,H,T,HD] f16 (rotated), Vt: [B,H,HD,T] f16
__global__ __launch_bounds__(256) void rope_pack_kernel(
    const float* __restrict__ qkv,
    _Float16* __restrict__ Qr, _Float16* __restrict__ Kr, _Float16* __restrict__ Vt) {
  int i = blockIdx.x * 256 + threadIdx.x;     // ((b*H+h)*T + t)*32 + pair
  if (i >= BB * HH * TT * 32) return;
  const int pair = i & 31;
  const int t    = (i >> 5) & (TT - 1);
  const int bh   = i >> 15;
  const int h    = bh % HH;
  const int b    = bh / HH;
  const size_t qrow = ((size_t)b * TT + t) * (3 * DD);
  const int base = h * HD + pair;
  float sn = 0.f, cs = 1.f;
  if (pair < 16) {               // rope_angles = int(0.5*64/2) = 16
    float ang = (float)t / powf(10000.0f, (2.0f * (float)pair) / 64.0f);
    sn = sinf(ang); cs = cosf(ang);
  }
  float q1 = qkv[qrow + base],            q2 = qkv[qrow + base + 32];
  float k1 = qkv[qrow + DD + base],       k2 = qkv[qrow + DD + base + 32];
  const size_t ob = ((size_t)bh * TT + t) * HD;
  Qr[ob + pair]      = (_Float16)(q1 * cs - q2 * sn);
  Qr[ob + pair + 32] = (_Float16)(q2 * cs + q1 * sn);
  Kr[ob + pair]      = (_Float16)(k1 * cs - k2 * sn);
  Kr[ob + pair + 32] = (_Float16)(k2 * cs + k1 * sn);
  const int d0 = pair * 2;
  Vt[((size_t)bh * HD + d0)     * TT + t] = (_Float16)qkv[qrow + 2 * DD + h * HD + d0];
  Vt[((size_t)bh * HD + d0 + 1) * TT + t] = (_Float16)qkv[qrow + 2 * DD + h * HD + d0 + 1];
}

// ---------------------------------------------------------------- generic f16 WMMA GEMM
// C[M,N] = A[M,K] @ W[K,N] with W given pre-transposed+padded: Bt [Npad,K] f16.
// Software-pipelined: next K-slab is loaded to registers while WMMAs consume LDS.
// FLAGS is compile-time: each fused epilogue variant is a separate specialization.
template <int FLAGS>
__global__ __launch_bounds__(256) void gemm_f16_kernel(
    const _Float16* __restrict__ A, const _Float16* __restrict__ Bt,
    const float* __restrict__ bias, const float* __restrict__ res,
    float* __restrict__ outF, _Float16* __restrict__ outH,
    int M, int N, int K) {
  __shared__ __attribute__((aligned(16))) _Float16 As[128][40];
  __shared__ __attribute__((aligned(16))) _Float16 Bs[128][40];

  const int tid  = threadIdx.x;
  const int lane = tid & 31;
  const int wave = tid >> 5;
  const int wm   = wave >> 1;           // 0..3 : 32-row strip
  const int wn   = wave & 1;            // 0..1 : 64-col strip
  const int bm   = blockIdx.y * 128;
  const int bn   = blockIdx.x * 128;
  const int ln   = lane & 15;
  const int kh   = lane >> 4;

  v8f acc[2][4];
  #pragma unroll
  for (int mi = 0; mi < 2; mi++)
    #pragma unroll
    for (int ni = 0; ni < 4; ni++)
      #pragma unroll
      for (int e = 0; e < 8; e++) acc[mi][ni][e] = 0.f;

  // staging map (identical for A and Bt): thread -> (tile row, 16-half chunk)
  const int r  = tid >> 1;              // 0..127
  const int c0 = (tid & 1) * 16;        // 0 or 16
  const _Float16* ga = A  + (size_t)(bm + r) * K + c0;
  const _Float16* gb = Bt + (size_t)(bn + r) * K + c0;

  // pipeline prologue: first K-slab into registers
  v8h a0 = *(const v8h*)ga, a1 = *(const v8h*)(ga + 8);
  v8h b0 = *(const v8h*)gb, b1 = *(const v8h*)(gb + 8);

  for (int k0 = 0; k0 < K; k0 += 32) {
    *(v8h*)&As[r][c0]     = a0;
    *(v8h*)&As[r][c0 + 8] = a1;
    *(v8h*)&Bs[r][c0]     = b0;
    *(v8h*)&Bs[r][c0 + 8] = b1;
    __syncthreads();

    if (k0 + 32 < K) {                  // issue next slab while WMMAs run
      ga += 32; gb += 32;
      a0 = *(const v8h*)ga; a1 = *(const v8h*)(ga + 8);
      b0 = *(const v8h*)gb; b1 = *(const v8h*)(gb + 8);
      if (k0 + 64 < K) {
        __builtin_prefetch(ga + 32, 0, 3);   // near-cache (WGP) prefetch
        __builtin_prefetch(gb + 32, 0, 3);
      }
    }

    v16h afr[2], bfr[4];
    #pragma unroll
    for (int mi = 0; mi < 2; mi++) {
      const int row = wm * 32 + mi * 16 + ln;
      const v8h x0 = *(const v8h*)&As[row][kh * 8];
      const v8h x1 = *(const v8h*)&As[row][16 + kh * 8];
      #pragma unroll
      for (int e = 0; e < 8; e++) { afr[mi][e] = x0[e]; afr[mi][8 + e] = x1[e]; }
    }
    #pragma unroll
    for (int ni = 0; ni < 4; ni++) {
      const int col = wn * 64 + ni * 16 + ln;
      const v8h x0 = *(const v8h*)&Bs[col][kh * 8];
      const v8h x1 = *(const v8h*)&Bs[col][16 + kh * 8];
      #pragma unroll
      for (int e = 0; e < 8; e++) { bfr[ni][e] = x0[e]; bfr[ni][8 + e] = x1[e]; }
    }
    #pragma unroll
    for (int mi = 0; mi < 2; mi++)
      #pragma unroll
      for (int ni = 0; ni < 4; ni++)
        acc[mi][ni] = __builtin_amdgcn_wmma_f32_16x16x32_f16(
            false, afr[mi], false, bfr[ni], (short)0, acc[mi][ni], false, false);
    __syncthreads();
  }

  // ---- fused epilogue (dead paths removed per specialization)
  #pragma unroll
  for (int mi = 0; mi < 2; mi++)
    #pragma unroll
    for (int ni = 0; ni < 4; ni++)
      #pragma unroll
      for (int r8 = 0; r8 < 8; r8++) {
        const int row = bm + wm * 32 + mi * 16 + r8 + 8 * kh;
        const int col = bn + wn * 64 + ni * 16 + ln;
        if (col < N) {
          float v = acc[mi][ni][r8];
          if (FLAGS & GEMM_BIAS) v += bias[col];
          if (FLAGS & GEMM_GELU) {
            float u = v;
            v = 0.5f * u * (1.f + tanhf(0.7978845608f * (u + 0.044715f * u * u * u)));
          }
          const size_t o = (size_t)row * N + col;
          if (FLAGS & GEMM_RES)  v += res[o];
          if (FLAGS & GEMM_WF32) outF[o] = v;
          if (FLAGS & GEMM_WF16) outH[o] = (_Float16)v;
        }
      }
}

// ---------------------------------------------------------------- flash attention (WMMA)
// One wave per (b,h, 16-query tile). Qr,Kr: [B,H,T,HD], Vt: [B,H,HD,T]; y: [B,T,D] f16.
__global__ __launch_bounds__(32) void attn_kernel(
    const _Float16* __restrict__ Qr, const _Float16* __restrict__ Kr,
    const _Float16* __restrict__ Vt, _Float16* __restrict__ y) {
  __shared__ __attribute__((aligned(16))) _Float16 Ps[16][40];
  const int lane = threadIdx.x;
  const int bid  = blockIdx.x;               // B*H*(T/16)
  const int qt   = bid % (TT / 16);
  const int bh   = bid / (TT / 16);
  const int h    = bh % HH;
  const int b    = bh / HH;
  const int q0   = qt * 16;
  const int ln   = lane & 15;
  const int kh   = lane >> 4;

  const _Float16* Qbh = Qr + (size_t)bh * TT * HD;
  const _Float16* Kbh = Kr + (size_t)bh * TT * HD;
  const _Float16* Vbh = Vt + (size_t)bh * HD * TT;

  // Q A-fragments (two K=32 chunks of HD=64)
  v16h aq[2];
  #pragma unroll
  for (int c = 0; c < 2; c++) {
    const _Float16* p = Qbh + (size_t)(q0 + ln) * HD + c * 32;
    const v8h a0 = *(const v8h*)(p + kh * 8);
    const v8h a1 = *(const v8h*)(p + 16 + kh * 8);
    #pragma unroll
    for (int e = 0; e < 8; e++) { aq[c][e] = a0[e]; aq[c][8 + e] = a1[e]; }
  }

  v8f accv[4];
  #pragma unroll
  for (int ni = 0; ni < 4; ni++)
    #pragma unroll
    for (int e = 0; e < 8; e++) accv[ni][e] = 0.f;
  float m8[8], l8[8];
  #pragma unroll
  for (int r = 0; r < 8; r++) { m8[r] = -3.0e38f; l8[r] = 0.f; }

  for (int kb = 0; kb < q0 + 16; kb += 32) {
    // ---- S = Q @ K^T   (16 x 32, as two 16x16 tiles)
    v8f s[2];
    #pragma unroll
    for (int j = 0; j < 2; j++) {
      #pragma unroll
      for (int e = 0; e < 8; e++) s[j][e] = 0.f;
      #pragma unroll
      for (int c = 0; c < 2; c++) {
        v16h bk;
        const _Float16* p = Kbh + (size_t)(kb + j * 16 + ln) * HD + c * 32;
        const v8h b0 = *(const v8h*)(p + kh * 8);
        const v8h b1 = *(const v8h*)(p + 16 + kh * 8);
        #pragma unroll
        for (int e = 0; e < 8; e++) { bk[e] = b0[e]; bk[8 + e] = b1[e]; }
        s[j] = __builtin_amdgcn_wmma_f32_16x16x32_f16(false, aq[c], false, bk,
                                                      (short)0, s[j], false, false);
      }
    }
    // ---- scale + causal mask
    float pm[2][8];
    #pragma unroll
    for (int j = 0; j < 2; j++)
      #pragma unroll
      for (int r = 0; r < 8; r++) {
        float vv = s[j][r] * 0.125f;                 // 1/sqrt(64)
        const int key = kb + j * 16 + ln;
        const int row = q0 + r + 8 * kh;
        pm[j][r] = (key > row) ? -3.0e38f : vv;
      }
    // ---- online softmax (row stats across 16-lane halves)
    float pr[2][8];
    #pragma unroll
    for (int r = 0; r < 8; r++) {
      float mloc = fmaxf(pm[0][r], pm[1][r]);
      #pragma unroll
      for (int o = 8; o > 0; o >>= 1) mloc = fmaxf(mloc, __shfl_xor(mloc, o, 32));
      const float mnew  = fmaxf(m8[r], mloc);
      const float scale = __expf(m8[r] - mnew);
      float sum = 0.f;
      #pragma unroll
      for (int j = 0; j < 2; j++) {
        float e = __expf(pm[j][r] - mnew);
        pr[j][r] = e;
        sum += e;
      }
      #pragma unroll
      for (int o = 8; o > 0; o >>= 1) sum += __shfl_xor(sum, o, 32);
      l8[r] = l8[r] * scale + sum;
      #pragma unroll
      for (int ni = 0; ni < 4; ni++) accv[ni][r] *= scale;
      m8[r] = mnew;
    }
    // ---- P: C-layout -> A-layout via LDS
    #pragma unroll
    for (int j = 0; j < 2; j++)
      #pragma unroll
      for (int r = 0; r < 8; r++)
        Ps[r + 8 * kh][j * 16 + ln] = (_Float16)pr[j][r];
    __syncthreads();
    v16h ap;
    {
      const _Float16* p = &Ps[ln][0];
      const v8h a0 = *(const v8h*)(p + kh * 8);
      const v8h a1 = *(const v8h*)(p + 16 + kh * 8);
      #pragma unroll
      for (int e = 0; e < 8; e++) { ap[e] = a0[e]; ap[8 + e] = a1[e]; }
    }
    __syncthreads();
    // ---- acc += P @ V
    #pragma unroll
    for (int ni = 0; ni < 4; ni++) {
      v16h bv;
      const _Float16* p = Vbh + (size_t)(ni * 16 + ln) * TT + kb;
      const v8h b0 = *(const v8h*)(p + kh * 8);
      const v8h b1 = *(const v8h*)(p + 16 + kh * 8);
      #pragma unroll
      for (int e = 0; e < 8; e++) { bv[e] = b0[e]; bv[8 + e] = b1[e]; }
      accv[ni] = __builtin_amdgcn_wmma_f32_16x16x32_f16(false, ap, false, bv,
                                                        (short)0, accv[ni], false, false);
    }
  }
  // ---- normalize + store y[b, t, h*HD + d] as f16
  #pragma unroll
  for (int ni = 0; ni < 4; ni++)
    #pragma unroll
    for (int r = 0; r < 8; r++) {
      const int row = q0 + r + 8 * kh;
      const float v = accv[ni][r] / l8[r];
      y[((size_t)b * TT + row) * DD + h * HD + ni * 16 + ln] = (_Float16)v;
    }
}

// ---------------------------------------------------------------- host
extern "C" void kernel_launch(void* const* d_in, const int* in_sizes, int n_in,
                              void* d_out, int out_size, void* d_ws, size_t ws_size,
                              hipStream_t stream) {
  (void)in_sizes; (void)n_in; (void)out_size; (void)ws_size;
  const int*   idx  = (const int*)  d_in[0];
  const float* wte  = (const float*)d_in[1];
  const float* ln1g = (const float*)d_in[2];
  const float* ln1b = (const float*)d_in[3];
  const float* Wqkv = (const float*)d_in[4];
  const float* bqkv = (const float*)d_in[5];
  const float* Wo   = (const float*)d_in[6];
  const float* bo   = (const float*)d_in[7];
  const float* ln2g = (const float*)d_in[8];
  const float* ln2b = (const float*)d_in[9];
  const float* Wfc  = (const float*)d_in[10];
  const float* bfc  = (const float*)d_in[11];
  const float* Wp   = (const float*)d_in[12];
  const float* bp   = (const float*)d_in[13];
  const float* lnfg = (const float*)d_in[14];
  const float* lnfb = (const float*)d_in[15];
  const float* Wlm  = (const float*)d_in[16];

  char* ws = (char*)d_ws;
  size_t off = 0;
  auto carve = [&](size_t bytes) -> char* {
    char* p = ws + off;
    off = (off + bytes + 255) & ~(size_t)255;
    return p;
  };
  const int M = BB * TT;                                   // 2048
  // transposed f16 weights (W^T: [N,K])
  _Float16* WqkvT = (_Float16*)carve((size_t)LL * 3 * DD * DD * 2);
  _Float16* WoT   = (_Float16*)carve((size_t)LL * DD * DD * 2);
  _Float16* WfcT  = (_Float16*)carve((size_t)LL * 4 * DD * DD * 2);
  _Float16* WpT   = (_Float16*)carve((size_t)LL * DD * 4 * DD * 2);
  _Float16* WlmT  = (_Float16*)carve((size_t)VP * DD * 2);
  float*    x     = (float*)   carve((size_t)M * DD * 4);
  _Float16* h1    = (_Float16*)carve((size_t)M * DD * 2);
  _Float16* h2    = (_Float16*)carve((size_t)M * DD * 2);
  float*    qkvb  = (float*)   carve((size_t)M * 3 * DD * 4);
  _Float16* Qr    = (_Float16*)carve((size_t)BB * HH * TT * HD * 2);
  _Float16* Kr    = (_Float16*)carve((size_t)BB * HH * TT * HD * 2);
  _Float16* Vt    = (_Float16*)carve((size_t)BB * HH * TT * HD * 2);
  _Float16* yb    = (_Float16*)carve((size_t)M * DD * 2);
  _Float16* hb    = (_Float16*)carve((size_t)M * 4 * DD * 2);
  _Float16* xf    = (_Float16*)carve((size_t)M * DD * 2);

  auto transp = [&](const float* s, _Float16* d, int K, int N, int Npad) {
    dim3 grid(Npad / 32, K / 32), blk(32, 8);
    transpose_f16_kernel<<<grid, blk, 0, stream>>>(s, d, K, N, Npad);
  };
  for (int l = 0; l < LL; l++) {
    transp(Wqkv + (size_t)l * DD * 3 * DD, WqkvT + (size_t)l * 3 * DD * DD, DD, 3 * DD, 3 * DD);
    transp(Wo   + (size_t)l * DD * DD,     WoT   + (size_t)l * DD * DD,     DD, DD,     DD);
    transp(Wfc  + (size_t)l * DD * 4 * DD, WfcT  + (size_t)l * 4 * DD * DD, DD, 4 * DD, 4 * DD);
    transp(Wp   + (size_t)l * 4 * DD * DD, WpT   + (size_t)l * DD * 4 * DD, 4 * DD, DD, DD);
  }
  transp(Wlm, WlmT, DD, VV, VP);

  embed_kernel<<<dim3((BB * TT * DD + 255) / 256), dim3(256), 0, stream>>>(idx, wte, x);

  for (int l = 0; l < LL; l++) {
    ln_dual_kernel<<<dim3(M), dim3(256), 0, stream>>>(
        x, ln1g + l * DD, ln1b + l * DD, ln2g + l * DD, ln2b + l * DD, h1, h2);
    // QKV projection: h1 @ Wqkv + bqkv -> qkvb (f32)
    gemm_f16_kernel<GEMM_BIAS | GEMM_WF32>
        <<<dim3((3 * DD) / 128, M / 128), dim3(256), 0, stream>>>(
        h1, WqkvT + (size_t)l * 3 * DD * DD, bqkv + (size_t)l * 3 * DD, nullptr,
        qkvb, nullptr, M, 3 * DD, DD);
    // RoPE + pack
    rope_pack_kernel<<<dim3((BB * HH * TT * 32) / 256), dim3(256), 0, stream>>>(qkvb, Qr, Kr, Vt);
    // flash attention
    attn_kernel<<<dim3(BB * HH * (TT / 16)), dim3(32), 0, stream>>>(Qr, Kr, Vt, yb);
    // output projection + residual: x += yb @ Wo + bo
    gemm_f16_kernel<GEMM_BIAS | GEMM_RES | GEMM_WF32>
        <<<dim3(DD / 128, M / 128), dim3(256), 0, stream>>>(
        yb, WoT + (size_t)l * DD * DD, bo + (size_t)l * DD, x,
        x, nullptr, M, DD, DD);
    // MLP fc + GELU -> hb (f16)
    gemm_f16_kernel<GEMM_BIAS | GEMM_GELU | GEMM_WF16>
        <<<dim3((4 * DD) / 128, M / 128), dim3(256), 0, stream>>>(
        h2, WfcT + (size_t)l * 4 * DD * DD, bfc + (size_t)l * 4 * DD, nullptr,
        nullptr, hb, M, 4 * DD, DD);
    // MLP proj + residual: x += hb @ Wp + bp
    gemm_f16_kernel<GEMM_BIAS | GEMM_RES | GEMM_WF32>
        <<<dim3(DD / 128, M / 128), dim3(256), 0, stream>>>(
        hb, WpT + (size_t)l * DD * 4 * DD, bp + (size_t)l * DD, x,
        x, nullptr, M, DD, 4 * DD);
  }

  // final LN -> f16
  ln_dual_kernel<<<dim3(M), dim3(256), 0, stream>>>(
      x, lnfg, lnfb, nullptr, nullptr, xf, nullptr);
  // LM head -> d_out (f32)
  gemm_f16_kernel<GEMM_WF32>
      <<<dim3(VP / 128, M / 128), dim3(256), 0, stream>>>(
      xf, WlmT, nullptr, nullptr, (float*)d_out, nullptr, M, VV, DD);
}
